// LSTM_20753281974857
// MI455X (gfx1250) — matrix operand order repaired and checked by reading, not compiled, not run on hardware
//
#include <hip/hip_runtime.h>
#include <hip/hip_bf16.h>
#include <math.h>

#define T_DIM 512
#define B_DIM 64
#define D_DIM 512
#define H_DIM 1024
#define G4H   (4 * H_DIM)

typedef __attribute__((ext_vector_type(16))) __bf16       v16bf;
typedef __attribute__((ext_vector_type(8)))  float        v8f;
typedef __attribute__((ext_vector_type(4)))  unsigned int u32x4;

union BfFrag { v16bf v; u32x4 q[2]; };

// A fragment (16x32 bf16, M x K), row-major source with leading dim ld.
// ISA layout: lane<16 -> row m0+lane, elems = K[k0..k0+7] ++ K[k0+16..k0+23]
//             lane>=16 -> row m0+lane-16, elems = K[k0+8..k0+15] ++ K[k0+24..k0+31]
__device__ inline v16bf load_a_frag(const __bf16* __restrict__ base, int ld,
                                    int m0, int k0, int lane) {
    const int half = lane >> 4;
    const int row  = m0 + (lane & 15);
    const __bf16* p = base + (size_t)row * ld + k0 + half * 8;
    BfFrag u;
    u.q[0] = *(const u32x4*)(p);
    u.q[1] = *(const u32x4*)(p + 16);
    return u.v;
}

// B fragment from pre-packed weights: pack[(nt*KT + kt)*512 + lane*16 + e]
// lane<16 -> col nt*16+lane, K[k0+0..7, k0+16..23]; lane>=16 -> col nt*16+lane-16, K[k0+8..15, k0+24..31]
__device__ inline v16bf load_b_frag(const __bf16* __restrict__ pack,
                                    int nt, int KT, int kt, int lane) {
    const __bf16* p = pack + (((size_t)nt * KT + kt) << 9) + lane * 16;
    BfFrag u;
    u.q[0] = *(const u32x4*)(p);
    u.q[1] = *(const u32x4*)(p + 8);
    return u.v;
}

__device__ inline v8f wmma_bf16(v16bf a, v16bf b, v8f c) {
    return __builtin_amdgcn_wmma_f32_16x16x32_bf16(
        /*neg_a=*/false, a, /*neg_b=*/false, b,
        /*c_mod=*/(short)0, c, /*reuse_a=*/false, /*reuse_b=*/false);
}

__device__ inline float sigmoidf_(float x) { return 1.0f / (1.0f + __expf(-x)); }

// ---------------- prep kernels ----------------

__global__ __launch_bounds__(256) void cvt_f32_bf16(const float* __restrict__ src,
                                                    __bf16* __restrict__ dst, int n) {
    int i = (blockIdx.x * 256 + threadIdx.x) * 4;
    if (i + 3 < n) {
        float4 f = *(const float4*)(src + i);
        dst[i + 0] = (__bf16)f.x;
        dst[i + 1] = (__bf16)f.y;
        dst[i + 2] = (__bf16)f.z;
        dst[i + 3] = (__bf16)f.w;
    }
}

// Pack 4 gate weight matrices (each [K][H] fp32, logical concat -> [K][4H]) into
// WMMA-B-fragment layout: out[nt][kt][lane][e], nt over 4H/16 tiles, kt over K/32.
__global__ __launch_bounds__(256) void pack_w4(const float* __restrict__ Wi,
                                               const float* __restrict__ Wf,
                                               const float* __restrict__ Wo,
                                               const float* __restrict__ Wc,
                                               __bf16* __restrict__ out, int K) {
    size_t idx = (size_t)blockIdx.x * 256 + threadIdx.x;   // one per packed element
    int    e    = (int)(idx & 15);
    int    lane = (int)((idx >> 4) & 31);
    size_t tile = idx >> 9;
    int    KT   = K >> 5;
    int    kt   = (int)(tile % KT);
    int    nt   = (int)(tile / KT);
    int    col  = nt * 16 + (lane & 15);          // global column in [0, 4H)
    int    k    = kt * 32 + ((lane >> 4) << 3) + (e & 7) + ((e >> 3) << 4);
    int    gate = col >> 10;                      // col / H
    int    h    = col & (H_DIM - 1);
    const float* W = (gate == 0) ? Wi : (gate == 1) ? Wf : (gate == 2) ? Wo : Wc;
    out[idx] = (__bf16)W[(size_t)k * H_DIM + h];
}

// ---------------- phase 1: Xg = X @ Wx + b  ([32768 x 512] x [512 x 4096]) ----------------

__global__ __launch_bounds__(256) void xg_gemm(const __bf16* __restrict__ Xbf,
                                               const __bf16* __restrict__ WxP,
                                               const float* __restrict__ bi,
                                               const float* __restrict__ bf_,
                                               const float* __restrict__ bo,
                                               const float* __restrict__ bc,
                                               float* __restrict__ Xg) {
    const int lane = threadIdx.x & 31;
    const int w    = threadIdx.x >> 5;
    const int wr   = w & 3, wc = w >> 2;           // 4x2 wave grid -> 128x64 block tile
    const int m0 = blockIdx.y * 128 + wr * 32;
    const int n0 = blockIdx.x * 64  + wc * 32;
    const int KT = D_DIM / 32;                     // 16 k-tiles

    v8f acc00 = {0,0,0,0,0,0,0,0}, acc01 = acc00, acc10 = acc00, acc11 = acc00;

    for (int kt = 0; kt < KT; ++kt) {
        const int k0 = kt * 32;
        v16bf a0 = load_a_frag(Xbf, D_DIM, m0,      k0, lane);
        v16bf a1 = load_a_frag(Xbf, D_DIM, m0 + 16, k0, lane);
        v16bf b0 = load_b_frag(WxP, (n0 >> 4),     KT, kt, lane);
        v16bf b1 = load_b_frag(WxP, (n0 >> 4) + 1, KT, kt, lane);
        acc00 = wmma_bf16(a0, b0, acc00);
        acc01 = wmma_bf16(a0, b1, acc01);
        acc10 = wmma_bf16(a1, b0, acc10);
        acc11 = wmma_bf16(a1, b1, acc11);
    }

    const int rbase = (lane < 16) ? 0 : 8;
    const int cl    = lane & 15;
#pragma unroll
    for (int ni = 0; ni < 2; ++ni) {
        const int col  = n0 + ni * 16 + cl;
        const int gate = col >> 10;
        const int h    = col & (H_DIM - 1);
        const float bias = ((gate == 0) ? bi : (gate == 1) ? bf_ : (gate == 2) ? bo : bc)[h];
#pragma unroll
        for (int mi = 0; mi < 2; ++mi) {
            const v8f& A = (mi == 0) ? (ni == 0 ? acc00 : acc01)
                                     : (ni == 0 ? acc10 : acc11);
#pragma unroll
            for (int v = 0; v < 8; ++v) {
                const int row = m0 + mi * 16 + v + rbase;
                Xg[(size_t)row * G4H + col] = A[v] + bias;
            }
        }
    }
}

// ---------------- phase 2: one fused step.  g = Xg[t] + H@Wh, gates, cell update ----------
// Grid: H/16 blocks. Block owns h-strip [h0, h0+16) for all 4 gates and all 64 batch rows.
// Waves: wr = rows 16*wr..16*wr+15, kh = K-half; kh=1 partials reduced via LDS.

__global__ __launch_bounds__(256) void lstm_step(const float* __restrict__ Xg,
                                                 const __bf16* __restrict__ WhP,
                                                 const __bf16* __restrict__ Hprev,
                                                 __bf16* __restrict__ Hnext,
                                                 float* __restrict__ out,  // [T][B][H]
                                                 float* __restrict__ Hf,
                                                 float* __restrict__ Cbuf,
                                                 int t) {
    __shared__ float red[4][4][8][32];   // [wr][gate][vreg][lane] = 16 KB
    const int lane = threadIdx.x & 31;
    const int w    = threadIdx.x >> 5;
    const int wr   = w & 3;
    const int kh   = w >> 2;
    const int h0   = blockIdx.x * 16;
    const int KT   = H_DIM / 32;         // 32 total k-tiles, 16 per half

    v8f acc[4] = { {0,0,0,0,0,0,0,0}, {0,0,0,0,0,0,0,0},
                   {0,0,0,0,0,0,0,0}, {0,0,0,0,0,0,0,0} };

    if (t > 0) {
        for (int kt = 0; kt < 16; ++kt) {
            const int ktG = kh * 16 + kt;
            const int k0  = ktG * 32;
            v16bf a = load_a_frag(Hprev, H_DIM, wr * 16, k0, lane);
#pragma unroll
            for (int g = 0; g < 4; ++g) {
                v16bf b = load_b_frag(WhP, g * (H_DIM / 16) + (h0 >> 4), KT, ktG, lane);
                acc[g] = wmma_bf16(a, b, acc[g]);
            }
        }
    }

    if (kh == 1) {
#pragma unroll
        for (int g = 0; g < 4; ++g)
#pragma unroll
            for (int v = 0; v < 8; ++v)
                red[wr][g][v][lane] = acc[g][v];
    }
    __syncthreads();

    if (kh == 0) {
#pragma unroll
        for (int g = 0; g < 4; ++g)
#pragma unroll
            for (int v = 0; v < 8; ++v)
                acc[g][v] += red[wr][g][v][lane];

        const int rbase = (lane < 16) ? 0 : 8;
        const int h     = h0 + (lane & 15);
#pragma unroll
        for (int v = 0; v < 8; ++v) {
            const int    b      = wr * 16 + v + rbase;
            const size_t xgBase = ((size_t)t * B_DIM + b) * G4H + h;
            float gi = acc[0][v] + Xg[xgBase];
            float gf = acc[1][v] + Xg[xgBase + H_DIM];
            float go = acc[2][v] + Xg[xgBase + 2 * H_DIM];
            float gc = acc[3][v] + Xg[xgBase + 3 * H_DIM];
            float I  = sigmoidf_(gi);
            float F  = sigmoidf_(gf);
            float O  = sigmoidf_(go);
            float Ct = tanhf(gc);
            float Cp = (t == 0) ? 0.0f : Cbuf[(size_t)b * H_DIM + h];
            float Cn = F * Cp + I * Ct;
            float Hn = O * tanhf(Cn);
            Cbuf[(size_t)b * H_DIM + h] = Cn;
            out[((size_t)t * B_DIM + b) * H_DIM + h] = Hn;
            Hnext[(size_t)b * H_DIM + h] = (__bf16)Hn;
            if (t == T_DIM - 1) Hf[(size_t)b * H_DIM + h] = Hn;
        }
    }
}

// ---------------- launch ----------------

extern "C" void kernel_launch(void* const* d_in, const int* in_sizes, int n_in,
                              void* d_out, int out_size, void* d_ws, size_t ws_size,
                              hipStream_t stream) {
    (void)in_sizes; (void)n_in; (void)out_size; (void)ws_size;
    const float* X   = (const float*)d_in[0];
    const float* Wxi = (const float*)d_in[1];
    const float* Whi = (const float*)d_in[2];
    const float* bi  = (const float*)d_in[3];
    const float* Wxf = (const float*)d_in[4];
    const float* Whf = (const float*)d_in[5];
    const float* bf_ = (const float*)d_in[6];
    const float* Wxo = (const float*)d_in[7];
    const float* Who = (const float*)d_in[8];
    const float* bo  = (const float*)d_in[9];
    const float* Wxc = (const float*)d_in[10];
    const float* Whc = (const float*)d_in[11];
    const float* bc  = (const float*)d_in[12];

    float* out = (float*)d_out;
    float* Hf  = out + (size_t)T_DIM * B_DIM * H_DIM;
    float* Cf  = Hf + (size_t)B_DIM * H_DIM;   // running cell state lives here

    char*   ws  = (char*)d_ws;
    float*  Xg  = (float*)ws;   ws += (size_t)T_DIM * B_DIM * G4H * sizeof(float);
    __bf16* Xbf = (__bf16*)ws;  ws += (size_t)T_DIM * B_DIM * D_DIM * sizeof(__bf16);
    __bf16* WxP = (__bf16*)ws;  ws += (size_t)D_DIM * G4H * sizeof(__bf16);
    __bf16* WhP = (__bf16*)ws;  ws += (size_t)H_DIM * G4H * sizeof(__bf16);
    __bf16* HA  = (__bf16*)ws;  ws += (size_t)B_DIM * H_DIM * sizeof(__bf16);
    __bf16* HB  = (__bf16*)ws;

    // 1) X -> bf16
    cvt_f32_bf16<<<(T_DIM * B_DIM * D_DIM) / 1024, 256, 0, stream>>>(
        X, Xbf, T_DIM * B_DIM * D_DIM);
    // 2) pack gate weights into WMMA B-fragment layout (bf16)
    pack_w4<<<((size_t)D_DIM * G4H) / 256, 256, 0, stream>>>(Wxi, Wxf, Wxo, Wxc, WxP, D_DIM);
    pack_w4<<<((size_t)H_DIM * G4H) / 256, 256, 0, stream>>>(Whi, Whf, Who, Whc, WhP, H_DIM);
    // 3) big input-projection GEMM (all T at once), bias fused
    xg_gemm<<<dim3(G4H / 64, (T_DIM * B_DIM) / 128), 256, 0, stream>>>(
        Xbf, WxP, bi, bf_, bo, bc, Xg);
    // 4) 512 sequential fused recurrent steps (Wh stays L2-resident)
    for (int t = 0; t < T_DIM; ++t) {
        const __bf16* hp = (t & 1) ? HB : HA;
        __bf16*       hn = (t & 1) ? HA : HB;
        lstm_step<<<H_DIM / 16, 256, 0, stream>>>(Xg, WhP, hp, hn, out, Hf, Cf, t);
    }
}
